// RecurrentActorCritic_38362647888446
// MI455X (gfx1250) — matrix operand order repaired and checked by reading, not compiled, not run on hardware
//
#include <hip/hip_runtime.h>
#include <math.h>

#define DEVI __device__ __forceinline__

typedef __attribute__((ext_vector_type(16))) __bf16 bf16x16;
typedef __attribute__((ext_vector_type(8)))  float  f32x8;
typedef __attribute__((__vector_size__(4 * sizeof(int)))) int v4i_;

namespace {
constexpr int kB = 256, kT = 512, kOBS = 512, kHID = 512, kRH = 1024, kA = 18;
constexpr int kBT  = kB * kT;          // 131072
constexpr int kRH3 = 3 * kRH;          // 3072
constexpr int LPK  = 72;               // LDS row pitch (bf16) for 64-wide K tiles; 144B rows, 16B-aligned
constexpr unsigned GRU_BLOCKS = 128;
}

union FragU { bf16x16 v; uint4 q[2]; };

DEVI unsigned short f2bf(float f) {
  union { float f; unsigned u; } c; c.f = f;
  unsigned r = c.u + 0x7FFFu + ((c.u >> 16) & 1u);   // round-to-nearest-even
  return (unsigned short)(r >> 16);
}
DEVI float bflo(unsigned u) { union { unsigned x; float f; } c; c.x = u << 16;        return c.f; }
DEVI float bfhi(unsigned u) { union { unsigned x; float f; } c; c.x = u & 0xffff0000u; return c.f; }
DEVI float bf2f(unsigned short h) { union { unsigned u; float f; } c; c.u = ((unsigned)h) << 16; return c.f; }
DEVI unsigned pack2(float a, float b) {
  return (unsigned)f2bf(a) | ((unsigned)f2bf(b) << 16);
}

// Fast activations: CDNA5 has V_TANH_F32 (TRANS op). One trans-op per gate.
DEVI float fast_tanh(float x) {
#if __has_builtin(__builtin_amdgcn_tanhf)
  return __builtin_amdgcn_tanhf(x);
#else
  float e = __expf(-2.0f * x);
  return (1.0f - e) * __builtin_amdgcn_rcpf(1.0f + e);
#endif
}
DEVI float fast_sigmoid(float x) { return fmaf(0.5f, fast_tanh(0.5f * x), 0.5f); }

// ---------------------------------------------------------------------------
// Async global->LDS staging (ASYNCcnt path): (int4 AS1*, int4 AS3*, imm, imm)
// ---------------------------------------------------------------------------
#if defined(__has_builtin)
#if __has_builtin(__builtin_amdgcn_global_load_async_to_lds_b128)
#define HAS_ASYNC 1
#endif
#endif
#ifndef HAS_ASYNC
#define HAS_ASYNC 0
#endif

DEVI void async_ld128(const unsigned short* g, unsigned short* l) {
#if HAS_ASYNC
  __builtin_amdgcn_global_load_async_to_lds_b128(
      (__attribute__((address_space(1))) v4i_*)g,
      (__attribute__((address_space(3))) v4i_*)l, 0, 0);
#else
  *(uint4*)l = *(const uint4*)g;
#endif
}
DEVI void wait_async0() {
#if HAS_ASYNC
#if __has_builtin(__builtin_amdgcn_s_wait_asynccnt)
  __builtin_amdgcn_s_wait_asynccnt(0);
#else
  asm volatile("s_wait_asynccnt 0" ::: "memory");
#endif
#endif
}

// A fragment (16x32 bf16, MxK) at column kbase: lane m=lane&15; lanes>=16 shift K by +8.
DEVI bf16x16 ldsA_frag(const unsigned short* s, int row, int kbase, int lane) {
  int o = (lane & 16) ? 8 : 0;
  const unsigned short* p = s + row * LPK + kbase + o;
  FragU f;
  f.q[0] = *(const uint4*)(p);
  f.q[1] = *(const uint4*)(p + 16);
  return f.v;
}
// B fragment (32x16 bf16, KxN) at column kbase: lane n=lane&15; lanes>=16 shift K by +8.
DEVI bf16x16 ldsB_frag(const unsigned short* s, int ncol, int kbase, int lane) {
  int o = (lane & 16) ? 8 : 0;
  const unsigned short* p = s + ncol * LPK + kbase + o;
  FragU f;
  f.q[0] = *(const uint4*)(p);
  f.q[1] = *(const uint4*)(p + 16);
  return f.v;
}
DEVI f32x8 wmma_bf16(bf16x16 a, bf16x16 b, f32x8 c) {
  return __builtin_amdgcn_wmma_f32_16x16x32_bf16(false, a, false, b, (short)0, c, false, false);
}

// Device-wide two-phase barrier (persistent GRU kernel).
DEVI void grid_sync(unsigned* bar, unsigned nblocks) {
  __syncthreads();
  if (threadIdx.x == 0) {
    volatile unsigned* gen = bar + 1;
    unsigned g = *gen;
    __threadfence();                       // release
    unsigned prev = atomicAdd(bar, 1u);
    if (prev == nblocks - 1) {
      bar[0] = 0;
      __threadfence();
      atomicAdd((unsigned*)(bar + 1), 1u);
    } else {
      while (*gen == g) { __builtin_amdgcn_s_sleep(2); }
    }
    __threadfence();                       // acquire (invalidates WGP cache)
  }
  __syncthreads();
}

// ---------------------------------------------------------------------------
// Weight convert (+ optional transpose to [N,K] row-major bf16)
// ---------------------------------------------------------------------------
__global__ void convert_kernel(const float* __restrict__ src,
                               unsigned short* __restrict__ dst,
                               int K, int N, int transpose) {
  int i = blockIdx.x * blockDim.x + threadIdx.x;
  if (i >= N * K) return;
  int n = i / K, k = i - n * K;
  float v = transpose ? src[(size_t)k * N + n] : src[i];
  dst[i] = f2bf(v);
}

// h buffers are stored PRE-MASKED for the consuming step (fp32 + bf16 copies).
__global__ void init_kernel(const float* __restrict__ hx,
                            const int* __restrict__ dones,
                            float* __restrict__ hf0,
                            unsigned short* __restrict__ hb0,
                            unsigned* __restrict__ bar) {
  int i = blockIdx.x * blockDim.x + threadIdx.x;
  if (i < 2) bar[i] = 0;
  if (i < kB * kRH) {
    int b = i / kRH;
    float v = (dones[b * kT] != 0) ? 0.0f : hx[i];
    hf0[i] = v;
    hb0[i] = f2bf(v);
  }
}

// ---------------------------------------------------------------------------
// Generic bf16 WMMA GEMM: Out(bf16) = epi(A * B^T + bias)
//   A: [M, K] (fp32 or bf16, row-major);  B: [N, K] bf16 (pre-transposed)
// Block tile 128(M) x 64(N), 8 waves, wave = 16(M) x 64(N), K-step 64.
// Double-buffered LDS + async DMA; 2-deep manual pipeline with SEPARATE
// accumulator sets per buffer parity (kills loop-carried register rotation).
// ---------------------------------------------------------------------------
template <bool A_F32, bool DO_TANH>
__global__ void __launch_bounds__(256)
wgemm_kernel(const void* __restrict__ Aptr, int lda,
             const unsigned short* __restrict__ Bw, int ldb,
             const float* __restrict__ bias,
             unsigned short* __restrict__ Out, int ldo, int K) {
  __shared__ __align__(16) unsigned short sA[2][128 * LPK];
  __shared__ __align__(16) unsigned short sB[2][64 * LPK];

  const int m0   = blockIdx.y * 128;
  const int n0   = blockIdx.x * 64;
  const int tid  = threadIdx.x;
  const int lane = tid & 31;
  const int wave = tid >> 5;
  const int mw   = wave * 16;
  const int lm   = lane & 15;

  f32x8 acc0[4] = {}, acc1[4] = {};

  auto stage = [&](int sel, int kk) {
    if (A_F32) {
      // 128 rows x 64 f32 -> bf16: each thread converts half a row (32 floats)
      int row = tid >> 1;
      int c0  = (tid & 1) * 32;
      const float* Af = (const float*)Aptr + (size_t)(m0 + row) * lda + kk + c0;
      unsigned* d32 = (unsigned*)&sA[sel][row * LPK + c0];
#pragma unroll
      for (int j = 0; j < 8; ++j) {
        float4 v = *(const float4*)(Af + j * 4);
        d32[2 * j + 0] = pack2(v.x, v.y);
        d32[2 * j + 1] = pack2(v.z, v.w);
      }
    } else {
      // A: 128 rows x 64 bf16 = 1024 x 16B transfers (4 per thread, uniform)
#pragma unroll
      for (int c = 0; c < 4; ++c) {
        int idx = tid + c * 256;
        int row = idx >> 3, q = idx & 7;
        async_ld128((const unsigned short*)Aptr + (size_t)(m0 + row) * lda + kk + q * 8,
                    &sA[sel][row * LPK + q * 8]);
      }
    }
    // B: 64 rows x 64 bf16 = 512 x 16B transfers (2 per thread, uniform)
#pragma unroll
    for (int c = 0; c < 2; ++c) {
      int idx = tid + c * 256;
      int n = idx >> 3, q = idx & 7;
      async_ld128(Bw + (size_t)(n0 + n) * ldb + kk + q * 8,
                  &sB[sel][n * LPK + q * 8]);
    }
  };

  auto compute = [&](int sel, f32x8* acc) {
    bf16x16 a0 = ldsA_frag(sA[sel], mw + lm, 0, lane);
    bf16x16 a1 = ldsA_frag(sA[sel], mw + lm, 32, lane);
#pragma unroll
    for (int nt = 0; nt < 4; ++nt) {
      bf16x16 b0 = ldsB_frag(sB[sel], nt * 16 + lm, 0, lane);
      acc[nt] = wmma_bf16(a0, b0, acc[nt]);
    }
#pragma unroll
    for (int nt = 0; nt < 4; ++nt) {
      bf16x16 b1 = ldsB_frag(sB[sel], nt * 16 + lm, 32, lane);
      acc[nt] = wmma_bf16(a1, b1, acc[nt]);
    }
  };

  const int NI = K / 64;                      // even (8 or 16)
  stage(0, 0);
  for (int ik = 0; ik < NI - 2; ik += 2) {    // uniform steady state
    wait_async0();
    __syncthreads();
    stage(1, (ik + 1) * 64);
    compute(0, acc0);
    wait_async0();
    __syncthreads();
    stage(0, (ik + 2) * 64);
    compute(1, acc1);
  }
  wait_async0();
  __syncthreads();
  stage(1, (NI - 1) * 64);
  compute(0, acc0);
  wait_async0();
  __syncthreads();
  compute(1, acc1);

  // epilogue: C layout VGPR i, lane l -> M = i + (l>=16?8:0), N = l&15
  const int cm = (lane >> 4) << 3;
#pragma unroll
  for (int nt = 0; nt < 4; ++nt) {
    int n = n0 + nt * 16 + lm;
    float b = bias[n];
    f32x8 s = acc0[nt] + acc1[nt];
#pragma unroll
    for (int i = 0; i < 8; ++i) {
      int m = m0 + mw + cm + i;
      float v = s[i] + b;
      if (DO_TANH) v = fast_tanh(v);
      Out[(size_t)m * ldo + n] = f2bf(v);
    }
  }
}

// ---------------------------------------------------------------------------
// Persistent GRU scan: 512 timesteps, device-wide barrier between steps.
// 128 blocks: 4 M-blocks (64 rows) x 32 N-blocks (32 cols of RH).
// Per step: 64x32x3-gate GEMM over K=1024 in K-steps of 64 (16 iterations,
// 6 WMMAs per sync, 1280 = 5x256 uniform DMA transfers per K-step).
// Separate accumulator sets per buffer parity; summed once per timestep.
// h is double-buffered, pre-masked, in fp32 (update) + bf16 (WMMA A) forms.
// ---------------------------------------------------------------------------
__global__ void __launch_bounds__(256)
gru_scan_kernel(const int* __restrict__ dones,
                const unsigned short* __restrict__ xproj,   // [BT,3RH] bf16 (b_ih folded)
                const unsigned short* __restrict__ Whh,     // [3RH,RH] bf16
                const float* __restrict__ bhh,
                float* __restrict__ hf0, float* __restrict__ hf1,
                unsigned short* __restrict__ hb0, unsigned short* __restrict__ hb1,
                unsigned short* __restrict__ gruout,        // [B,T,RH] bf16
                unsigned* __restrict__ bar) {
  __shared__ __align__(16) unsigned short sA[2][64 * LPK];
  __shared__ __align__(16) unsigned short sB[2][3][32 * LPK];

  const int tid  = threadIdx.x;
  const int lane = tid & 31;
  const int wave = tid >> 5;
  const int m0 = (blockIdx.x & 3) * 64;
  const int n0 = (blockIdx.x >> 2) * 32;
  const int wm = (wave & 3) * 16;
  const int wn = (wave >> 2) * 16;
  const int lm = lane & 15;

  float*          fb[2] = {hf0, hf1};
  unsigned short* bb[2] = {hb0, hb1};

  // gate biases are per-column: invariant across t, hoisted out of the scan
  const int cm = (lane >> 4) << 3;
  const int n  = n0 + wn + lm;
  const float bR = bhh[n], bZ = bhh[kRH + n], bN = bhh[2 * kRH + n];

  const int r8 = tid >> 3, q8 = tid & 7;

  f32x8 aR0, aZ0, aN0, aR1, aZ1, aN1;

  auto stage = [&](int sel, int kk, const unsigned short* curB) {
    // A: 64 rows x 64 bf16 = 512 x 16B (2 per thread)
#pragma unroll
    for (int c = 0; c < 2; ++c) {
      int idx = tid + c * 256;
      int row = idx >> 3, q = idx & 7;
      async_ld128(curB + (size_t)(m0 + row) * kRH + kk + q * 8,
                  &sA[sel][row * LPK + q * 8]);
    }
    // B: per gate 32 rows x 64 bf16 = 256 x 16B (1 per thread, uniform)
#pragma unroll
    for (int g = 0; g < 3; ++g) {
      async_ld128(Whh + (size_t)(g * kRH + n0 + r8) * kRH + kk + q8 * 8,
                  &sB[sel][g][r8 * LPK + q8 * 8]);
    }
  };

  auto compute = [&](int sel, f32x8& r, f32x8& z, f32x8& nn) {
    bf16x16 a0 = ldsA_frag(sA[sel], wm + lm, 0, lane);
    bf16x16 a1 = ldsA_frag(sA[sel], wm + lm, 32, lane);
    bf16x16 b;
    b = ldsB_frag(sB[sel][0], wn + lm, 0, lane);  r  = wmma_bf16(a0, b, r);
    b = ldsB_frag(sB[sel][1], wn + lm, 0, lane);  z  = wmma_bf16(a0, b, z);
    b = ldsB_frag(sB[sel][2], wn + lm, 0, lane);  nn = wmma_bf16(a0, b, nn);
    b = ldsB_frag(sB[sel][0], wn + lm, 32, lane); r  = wmma_bf16(a1, b, r);
    b = ldsB_frag(sB[sel][1], wn + lm, 32, lane); z  = wmma_bf16(a1, b, z);
    b = ldsB_frag(sB[sel][2], wn + lm, 32, lane); nn = wmma_bf16(a1, b, nn);
  };

  constexpr int NI = kRH / 64;   // 16 (even)

  for (int t = 0; t < kT; ++t) {
    const float*          curF = fb[t & 1];
    float*                nxtF = fb[(t + 1) & 1];
    const unsigned short* curB = bb[t & 1];
    unsigned short*       nxtB = bb[(t + 1) & 1];

    aR0 = (f32x8){}; aZ0 = (f32x8){}; aN0 = (f32x8){};
    aR1 = (f32x8){}; aZ1 = (f32x8){}; aN1 = (f32x8){};

    stage(0, 0, curB);
    for (int ik = 0; ik < NI - 2; ik += 2) {    // uniform steady state
      wait_async0();
      __syncthreads();
      stage(1, (ik + 1) * 64, curB);
      compute(0, aR0, aZ0, aN0);
      wait_async0();
      __syncthreads();
      stage(0, (ik + 2) * 64, curB);
      compute(1, aR1, aZ1, aN1);
    }
    wait_async0();
    __syncthreads();
    stage(1, (NI - 1) * 64, curB);
    compute(0, aR0, aZ0, aN0);
    wait_async0();
    __syncthreads();
    compute(1, aR1, aZ1, aN1);

    const f32x8 aR = aR0 + aR1;
    const f32x8 aZ = aZ0 + aZ1;
    const f32x8 aN = aN0 + aN1;

    // fused gate epilogue on the 16x16 C fragment (all fp32, TRANS-op gates)
#pragma unroll
    for (int i = 0; i < 8; ++i) {
      int m = m0 + wm + cm + i;
      float hold = curF[(size_t)m * kRH + n];
      size_t xrow = ((size_t)m * kT + t) * (size_t)kRH3;
      float xr = bf2f(xproj[xrow + n]);
      float xz = bf2f(xproj[xrow + kRH + n]);
      float xn = bf2f(xproj[xrow + 2 * kRH + n]);
      float r  = fast_sigmoid(xr + aR[i] + bR);
      float z  = fast_sigmoid(xz + aZ[i] + bZ);
      float nn = fast_tanh(xn + r * (aN[i] + bN));
      float hnew = (1.0f - z) * nn + z * hold;
      // pre-mask for the NEXT step's consumption
      float nm = 1.0f;
      if (t + 1 < kT) nm = (dones[m * kT + t + 1] != 0) ? 0.0f : 1.0f;
      float hm = hnew * nm;
      nxtF[(size_t)m * kRH + n] = hm;
      nxtB[(size_t)m * kRH + n] = f2bf(hm);
      gruout[((size_t)m * kT + t) * kRH + n] = f2bf(hnew);
    }
    grid_sync(bar, GRU_BLOCKS);
  }
}

// ---------------------------------------------------------------------------
// Tiny heads (N=18 / N=1): VALU dot products on packed bf16 pairs.
// ---------------------------------------------------------------------------
__global__ void logits_kernel(const unsigned short* __restrict__ Ha,
                              const float* __restrict__ Wa2,
                              const float* __restrict__ ba2,
                              float* __restrict__ out) {
  int i = blockIdx.x * blockDim.x + threadIdx.x;
  if (i >= kBT * kA) return;
  int row = i / kA, a = i - row * kA;
  float acc = ba2[a];
  const unsigned* h = (const unsigned*)(Ha + (size_t)row * kHID);
  for (int k2 = 0; k2 < kHID / 2; ++k2) {
    unsigned u = h[k2];
    acc += bflo(u) * Wa2[(2 * k2) * kA + a] + bfhi(u) * Wa2[(2 * k2 + 1) * kA + a];
  }
  out[i] = acc;
}

__global__ void values_kernel(const unsigned short* __restrict__ Hc,
                              const float* __restrict__ Wc2,
                              const float* __restrict__ bc2,
                              float* __restrict__ out) {
  int row = blockIdx.x * blockDim.x + threadIdx.x;
  if (row >= kBT) return;
  float acc = bc2[0];
  const unsigned* h = (const unsigned*)(Hc + (size_t)row * kHID);
  for (int k2 = 0; k2 < kHID / 2; ++k2) {
    unsigned u = h[k2];
    acc += bflo(u) * Wc2[2 * k2] + bfhi(u) * Wc2[2 * k2 + 1];
  }
  out[row] = acc;
}

__global__ void copy_hlast_kernel(const float* __restrict__ src,
                                  float* __restrict__ dst) {
  int i = blockIdx.x * blockDim.x + threadIdx.x;
  if (i < kB * kRH) dst[i] = src[i];
}

// ---------------------------------------------------------------------------
extern "C" void kernel_launch(void* const* d_in, const int* in_sizes, int n_in,
                              void* d_out, int out_size, void* d_ws, size_t ws_size,
                              hipStream_t stream) {
  (void)in_sizes; (void)n_in; (void)out_size; (void)ws_size;

  const float* x    = (const float*)d_in[0];
  const float* hx   = (const float*)d_in[1];
  const int*   dn   = (const int*)d_in[2];
  const float* Wb   = (const float*)d_in[3];
  const float* bb   = (const float*)d_in[4];
  const float* Wih  = (const float*)d_in[5];
  const float* Whh  = (const float*)d_in[6];
  const float* bih  = (const float*)d_in[7];
  const float* bhh  = (const float*)d_in[8];
  const float* Wa1  = (const float*)d_in[9];
  const float* ba1  = (const float*)d_in[10];
  const float* Wa2  = (const float*)d_in[11];
  const float* ba2  = (const float*)d_in[12];
  const float* Wc1  = (const float*)d_in[13];
  const float* bc1  = (const float*)d_in[14];
  const float* Wc2  = (const float*)d_in[15];
  const float* bc2  = (const float*)d_in[16];

  char* ws = (char*)d_ws;
  size_t off = 0;
  auto alloc = [&](size_t bytes) -> void* {
    void* p = ws + off;
    off = (off + bytes + 255) & ~(size_t)255;
    return p;
  };
  unsigned short* Wbt  = (unsigned short*)alloc((size_t)kOBS * kHID * 2);  // [HID, OBS]
  unsigned short* Wihb = (unsigned short*)alloc((size_t)kRH3 * kHID * 2);  // [3RH, HID]
  unsigned short* Whhb = (unsigned short*)alloc((size_t)kRH3 * kRH  * 2);  // [3RH, RH]
  unsigned short* Wa1t = (unsigned short*)alloc((size_t)kRH  * kHID * 2);  // [HID, RH]
  unsigned short* Wc1t = (unsigned short*)alloc((size_t)kRH  * kHID * 2);  // [HID, RH]
  float*          hf0  = (float*)alloc((size_t)kB * kRH * 4);
  float*          hf1  = (float*)alloc((size_t)kB * kRH * 4);
  unsigned short* hb0  = (unsigned short*)alloc((size_t)kB * kRH * 2);
  unsigned short* hb1  = (unsigned short*)alloc((size_t)kB * kRH * 2);
  unsigned*       bar  = (unsigned*)alloc(256);
  unsigned short* feat = (unsigned short*)alloc((size_t)kBT * kHID * 2);   // reused as Ha
  unsigned short* gru  = (unsigned short*)alloc((size_t)kBT * kRH  * 2);
  unsigned short* xprj = (unsigned short*)alloc((size_t)kBT * kRH3 * 2);   // reused as Hc
  unsigned short* Ha = feat;
  unsigned short* Hc = xprj;

  float* out_logits = (float*)d_out;
  float* out_values = out_logits + (size_t)kBT * kA;
  float* out_hlast  = out_values + (size_t)kBT;

  const int CT = 256;
  convert_kernel<<<(kOBS * kHID + CT - 1) / CT, CT, 0, stream>>>(Wb,  Wbt,  kOBS, kHID, 1);
  convert_kernel<<<(kRH3 * kHID + CT - 1) / CT, CT, 0, stream>>>(Wih, Wihb, kHID, kRH3, 0);
  convert_kernel<<<(kRH3 * kRH  + CT - 1) / CT, CT, 0, stream>>>(Whh, Whhb, kRH,  kRH3, 0);
  convert_kernel<<<(kRH  * kHID + CT - 1) / CT, CT, 0, stream>>>(Wa1, Wa1t, kRH,  kHID, 1);
  convert_kernel<<<(kRH  * kHID + CT - 1) / CT, CT, 0, stream>>>(Wc1, Wc1t, kRH,  kHID, 1);
  init_kernel<<<(kB * kRH + CT - 1) / CT, CT, 0, stream>>>(hx, dn, hf0, hb0, bar);

  // feat = tanh(x @ W_base + b_base)          [BT, HID]
  wgemm_kernel<true, true><<<dim3(kHID / 64, kBT / 128), 256, 0, stream>>>(
      x, kOBS, Wbt, kOBS, bb, feat, kHID, kOBS);
  // x_proj = feat @ W_ih^T + b_ih             [BT, 3RH]
  wgemm_kernel<false, false><<<dim3(kRH3 / 64, kBT / 128), 256, 0, stream>>>(
      feat, kHID, Wihb, kHID, bih, xprj, kRH3, kHID);
  // persistent GRU scan
  gru_scan_kernel<<<GRU_BLOCKS, 256, 0, stream>>>(dn, xprj, Whhb, bhh,
                                                  hf0, hf1, hb0, hb1, gru, bar);
  // actor / critic hidden layers
  wgemm_kernel<false, true><<<dim3(kHID / 64, kBT / 128), 256, 0, stream>>>(
      gru, kRH, Wa1t, kRH, ba1, Ha, kHID, kRH);
  wgemm_kernel<false, true><<<dim3(kHID / 64, kBT / 128), 256, 0, stream>>>(
      gru, kRH, Wc1t, kRH, bc1, Hc, kHID, kRH);
  // tiny output heads + h_last
  logits_kernel<<<(kBT * kA + CT - 1) / CT, CT, 0, stream>>>(Ha, Wa2, ba2, out_logits);
  values_kernel<<<(kBT + CT - 1) / CT, CT, 0, stream>>>(Hc, Wc2, bc2, out_values);
  copy_hlast_kernel<<<(kB * kRH + CT - 1) / CT, CT, 0, stream>>>(hf0, out_hlast);
}
